// MaskedConv1d_73512660238790
// MI455X (gfx1250) — 8-bit matrix operand order repaired and checked by reading, compiled, not run on hardware
//
#include <hip/hip_runtime.h>

typedef __attribute__((ext_vector_type(8))) int v8i;
typedef __attribute__((ext_vector_type(4))) float f4;
typedef __attribute__((ext_vector_type(4))) unsigned u4;

#define B_DIM   32
#define C_DIM   1024
#define L_DIM   2048
#define HEADS   128
#define KSZ     33
#define ROWS    (B_DIM * C_DIM)
#define OUT_BASE ((size_t)B_DIM * C_DIM * L_DIM)
#define LROW    2112

__global__ void MaskedConv1d_init_kernel(unsigned* gmax) {
    if (threadIdx.x == 0) *gmax = 0u;
}

__global__ __launch_bounds__(256) void MaskedConv1d_absmax_kernel(
        const float* __restrict__ x, const int* __restrict__ lens, unsigned* __restrict__ gmax) {
    __shared__ float red[256];
    const int N4 = (int)(OUT_BASE / 4);
    float m = 0.0f;
    for (int qi = blockIdx.x * blockDim.x + threadIdx.x; qi < N4; qi += gridDim.x * blockDim.x) {
        int q = (N4 - 1) - qi;
        f4 v = ((const f4*)x)[q];
        int e = q << 2;
        int b = e >> 21;
        int t = e & (L_DIM - 1);
        int len = lens[b];
        if (t + 3 < len) {
            m = fmaxf(m, fmaxf(fmaxf(fabsf(v.x), fabsf(v.y)), fmaxf(fabsf(v.z), fabsf(v.w))));
        } else {
            if (t     < len) m = fmaxf(m, fabsf(v.x));
            if (t + 1 < len) m = fmaxf(m, fabsf(v.y));
            if (t + 2 < len) m = fmaxf(m, fabsf(v.z));
            if (t + 3 < len) m = fmaxf(m, fabsf(v.w));
        }
    }
    red[threadIdx.x] = m;
    __syncthreads();
    for (int s = 128; s > 0; s >>= 1) {
        if (threadIdx.x < s) red[threadIdx.x] = fmaxf(red[threadIdx.x], red[threadIdx.x + s]);
        __syncthreads();
    }
    if (threadIdx.x == 0) atomicMax(gmax, __float_as_uint(red[0]));
}

__global__ void MaskedConv1d_prep_kernel(
        const float* __restrict__ w, const int* __restrict__ lens,
        const unsigned* __restrict__ gmax,
        float* __restrict__ out, float* __restrict__ ws_sa,
        float* __restrict__ ws_scale, unsigned char* __restrict__ ws_A) {
    int h = threadIdx.x;
    float amax = __uint_as_float(*gmax);
    float s_a = amax / 127.0f;
    if (h == 0) *ws_sa = s_a;

    float wmax = 0.0f;
    for (int k = 0; k < KSZ; ++k) wmax = fmaxf(wmax, fabsf(w[h * KSZ + k]));
    float s_w = wmax / 127.0f;
    float inv_sw = (s_w > 0.0f) ? 1.0f / s_w : 0.0f;

    int qw[KSZ];
    for (int k = 0; k < KSZ; ++k) {
        float r = __builtin_rintf(w[h * KSZ + k] * inv_sw);
        qw[k] = (int)fmaxf(-128.0f, fminf(127.0f, r));
    }

    float osc = s_a * s_w;
    ws_scale[h] = osc;
    out[OUT_BASE + 32 + h] = osc;
    if (h < 32) out[OUT_BASE + h] = (float)lens[h];

    unsigned* dst = (unsigned*)(ws_A + (size_t)h * 1024);
    for (int lane = 0; lane < 32; ++lane) {
        int m = lane & 15;
        int kofs = (lane < 16) ? 0 : 16;
        for (int v = 0; v < 8; ++v) {
            int kstart = (v >> 2) * 32 + kofs + (v & 3) * 4;
            unsigned u = 0;
            for (int j = 0; j < 4; ++j) {
                int d = (kstart + j) - m;
                int q = (d >= 0 && d <= 32) ? (qw[d] & 255) : 0;
                u |= (unsigned)q << (8 * j);
            }
            dst[lane * 8 + v] = u;
        }
    }
}

__global__ __launch_bounds__(256) void MaskedConv1d_conv_kernel(
        const float* __restrict__ x, const int* __restrict__ lens,
        const float* __restrict__ ws_sa, const float* __restrict__ ws_scale,
        const unsigned char* __restrict__ ws_A, float* __restrict__ out) {
    __shared__ __attribute__((aligned(16))) unsigned char lds[8 * LROW];

    const int lane = threadIdx.x & 31;
    const int wid  = threadIdx.x >> 5;
    const int row  = blockIdx.x * 8 + wid;
    const int b    = row >> 10;
    const int c    = row & (C_DIM - 1);
    const int h    = c & (HEADS - 1);

    const int   len   = lens[b];
    const float s_a   = *ws_sa;
    const float inv   = (s_a > 0.0f) ? 1.0f / s_a : 0.0f;
    const float scale = ws_scale[h];

    const u4* Ap = (const u4*)(ws_A + (size_t)h * 1024 + lane * 32);
    u4 a0 = Ap[0], a1 = Ap[1];
    v8i A;
    A[0] = (int)a0.x; A[1] = (int)a0.y; A[2] = (int)a0.z; A[3] = (int)a0.w;
    A[4] = (int)a1.x; A[5] = (int)a1.y; A[6] = (int)a1.z; A[7] = (int)a1.w;

    unsigned char* L = lds + wid * LROW;
    if (lane < 4) *(unsigned*)(L + lane * 4) = 0u;
    if (lane < 8) *(unsigned*)(L + 2064 + lane * 4) = 0u;

    const float* xr = x + (size_t)row * L_DIM;
    #pragma unroll
    for (int i = 0; i < 16; ++i) {
        int idx = i * 128 + lane * 4;
        f4 v = __builtin_nontemporal_load((const f4*)(xr + idx));
        float v0 = (idx     < len) ? v.x : 0.0f;
        float v1 = (idx + 1 < len) ? v.y : 0.0f;
        float v2 = (idx + 2 < len) ? v.z : 0.0f;
        float v3 = (idx + 3 < len) ? v.w : 0.0f;
        int q0 = (int)fmaxf(-128.0f, fminf(127.0f, __builtin_rintf(v0 * inv)));
        int q1 = (int)fmaxf(-128.0f, fminf(127.0f, __builtin_rintf(v1 * inv)));
        int q2 = (int)fmaxf(-128.0f, fminf(127.0f, __builtin_rintf(v2 * inv)));
        int q3 = (int)fmaxf(-128.0f, fminf(127.0f, __builtin_rintf(v3 * inv)));
        unsigned packed = (unsigned)(q0 & 255) | ((unsigned)(q1 & 255) << 8) |
                          ((unsigned)(q2 & 255) << 16) | ((unsigned)(q3 & 255) << 24);
        *(unsigned*)(L + 16 + idx) = packed;
    }

    float* orow = out + (size_t)row * L_DIM;
    const int p    = lane & 15;
    const int koff = (lane < 16) ? 0 : 16;
    const int m0   = (lane < 16) ? 0 : 8;

    #pragma unroll
    for (int tile = 0; tile < 8; ++tile) {
        int t0 = tile * 256;
        const u4* Bp = (const u4*)(L + t0 + 16 * p + koff);
        u4 b0 = Bp[0];
        u4 b1 = Bp[2];
        v8i Bm;
        Bm[0] = (int)b0.x; Bm[1] = (int)b0.y; Bm[2] = (int)b0.z; Bm[3] = (int)b0.w;
        Bm[4] = (int)b1.x; Bm[5] = (int)b1.y; Bm[6] = (int)b1.z; Bm[7] = (int)b1.w;
        v8i Cz = {0, 0, 0, 0, 0, 0, 0, 0};
        v8i D = __builtin_amdgcn_wmma_i32_16x16x64_iu8(true, A, true, Bm, Cz, false, false);
        f4 f0, f1;
        f0.x = scale * (float)D[0]; f0.y = scale * (float)D[1];
        f0.z = scale * (float)D[2]; f0.w = scale * (float)D[3];
        f1.x = scale * (float)D[4]; f1.y = scale * (float)D[5];
        f1.z = scale * (float)D[6]; f1.w = scale * (float)D[7];
        __builtin_nontemporal_store(f0, (f4*)(orow + t0 + 16 * p + m0));
        __builtin_nontemporal_store(f1, (f4*)(orow + t0 + 16 * p + m0 + 4));
    }
}

extern "C" void kernel_launch(void* const* d_in, const int* in_sizes, int n_in,
                              void* d_out, int out_size, void* d_ws, size_t ws_size,
                              hipStream_t stream) {
    const float* x    = (const float*)d_in[0];
    const int*   lens = (const int*)d_in[1];
    const float* w    = (const float*)d_in[2];
    float* out = (float*)d_out;

    unsigned*      ws_max   = (unsigned*)d_ws;
    float*         ws_sa    = (float*)((char*)d_ws + 64);
    float*         ws_scale = (float*)((char*)d_ws + 128);
    unsigned char* ws_A     = (unsigned char*)d_ws + 1024;

    MaskedConv1d_init_kernel<<<1, 32, 0, stream>>>(ws_max);
    MaskedConv1d_absmax_kernel<<<4096, 256, 0, stream>>>(x, lens, ws_max);
    MaskedConv1d_prep_kernel<<<1, 128, 0, stream>>>(w, lens, ws_max, out, ws_sa, ws_scale, ws_A);
    MaskedConv1d_conv_kernel<<<ROWS / 8, 256, 0, stream>>>(x, lens, ws_sa, ws_scale, ws_A, out);
}
